// FastQuantumV2_77129022702014
// MI455X (gfx1250) — matrix operand order repaired and checked by reading, not compiled, not run on hardware
//
#include <hip/hip_runtime.h>
#include <stdint.h>

// ---------------- problem constants (match reference) ----------------
#define NN   50000      // nodes  (== 3125 * 16, so no row-edge guards needed)
#define EE   300000     // edges (before self-loops)
#define EPQ  (EE + NN)  // edges incl. self-loops = 350000
#define GG   128        // graphs
#define HH   4          // heads
#define CC   64         // channels per head
#define INF_ 32         // input features
#define HCC  256        // H*C

// ---------------- CDNA5 vector types ----------------
typedef __attribute__((ext_vector_type(16))) __bf16  v16bf;
typedef __attribute__((ext_vector_type(8)))  __bf16  v8bf;
typedef __attribute__((ext_vector_type(8)))  float   v8f;
typedef __attribute__((ext_vector_type(4)))  unsigned int u32x4;
typedef __attribute__((ext_vector_type(8)))  int     i32x8;

#define ENABLE_TDM 1

// f32 -> bf16 round-to-nearest-even
__device__ inline __bf16 f2bf(float f) {
    union { float f; unsigned u; } v; v.f = f;
    unsigned r = (v.u + 0x7FFFu + ((v.u >> 16) & 1u)) >> 16;
    unsigned short s = (unsigned short)r;
    __bf16 o; __builtin_memcpy(&o, &s, 2); return o;
}

// float atomic-max via sign-aware integer atomics (init with -inf)
__device__ inline void atomicMaxF(float* addr, float val) {
    if (val >= 0.0f) atomicMax((int*)addr, __float_as_int(val));
    else             atomicMin((unsigned int*)addr, (unsigned int)__float_as_int(val));
}

// =====================================================================
// Node-feature GEMM:  Hout[N, 256] = X[N, DIN] @ W[DIN, 256]
//   X  : bf16 row-major [N, DIN]      (pre-converted activations)
//   Wt : bf16 col-major [256, DIN]    (pre-transposed weights, L2-resident)
// Block = 256 threads = 8 waves; block tile 16 rows x 128 cols (grid.y=2).
// A-tile (16 x DIN bf16) is DMA'd into LDS by the Tensor Data Mover.
// Inner loop per wave: 2x ds_load_b128 (A frag) + 2x global_load_b128
// (contiguous B frag) + v_wmma_f32_16x16x32_bf16, fully unrolled over K.
// Epilogue: stage 16x128 f32 tile in LDS -> coalesced global_store_b128.
// =====================================================================
template<int DIN>
__global__ void __launch_bounds__(256)
k_lin_wmma(const __bf16* __restrict__ X, const __bf16* __restrict__ Wt,
           float* __restrict__ Hout)
{
    __shared__ __attribute__((aligned(16))) __bf16 Atile[16 * DIN];
    __shared__ __attribute__((aligned(16))) float  Ctile[16 * 128];
    const int wave    = threadIdx.x >> 5;
    const int lane    = threadIdx.x & 31;
    const int rowbase = blockIdx.x * 16;
    const int colbase = blockIdx.y * 128 + wave * 16;

#if ENABLE_TDM
    if (wave == 0) {
        // ---- Tensor DMA descriptor (D#), groups 0 & 1, 2D bf16 tile ----
        unsigned long long gaddr = (unsigned long long)(uintptr_t)(const void*)
                                   (X + (long long)rowbase * DIN);
        unsigned int ldsoff = (unsigned int)(uintptr_t)(void*)&Atile[0];
        u32x4 g0; i32x8 g1;
        g0[0] = 1u;                                            // count=1, user mode
        g0[1] = ldsoff;                                        // lds_addr
        g0[2] = (unsigned int)(gaddr & 0xFFFFFFFFull);         // global_addr[31:0]
        g0[3] = (unsigned int)((gaddr >> 32) & 0x01FFFFFFull)  // global_addr[56:32]
              | (2u << 30);                                    // type = 2 ("image")
        g1[0] = (int)(1u << 16);                               // data_size = 2B
        g1[1] = (int)(((unsigned)DIN & 0xFFFFu) << 16);        // tensor_dim0 lo16
        g1[2] = (int)(16u << 16);                              // dim0 hi | tensor_dim1=16
        g1[3] = (int)(((unsigned)DIN) << 16);                  // dim1 hi | tile_dim0=DIN
        g1[4] = 16;                                            // tile_dim1 = 16 rows
        g1[5] = DIN;                                           // tensor_dim0_stride
        g1[6] = 0; g1[7] = 0;
        asm volatile("tensor_load_to_lds %0, %1" :: "s"(g0), "s"(g1) : "memory");
        asm volatile("s_wait_tensorcnt 0x0" ::: "memory");
    }
    __syncthreads();
#else
    for (int i = threadIdx.x; i < 16 * DIN; i += 256)
        Atile[i] = X[(long long)rowbase * DIN + i];
    __syncthreads();
#endif

    __builtin_prefetch(&Wt[(long long)colbase * DIN], 0, 0);   // global_prefetch_b8

    v8f acc = {};
    const int half = lane >> 4;
    const int mrow = lane & 15;
    const int ncol = lane & 15;
    const __bf16* arow = &Atile[mrow * DIN];
    const __bf16* bcol = &Wt[(long long)(colbase + ncol) * DIN];

#pragma unroll
    for (int k0 = 0; k0 < DIN; k0 += 32) {
        // A 16x32 bf16 fragment: per half, contiguous K 16*half .. 16*half+15 (matches B)
        v16bf a = *(const v16bf*)&arow[k0 + half * 16];
        // B 32x16 fragment: K split lanes 0-15 / 16-31 -> contiguous 16 bf16 of Wt column
        v16bf b = *(const v16bf*)&bcol[k0 + half * 16];
        acc = __builtin_amdgcn_wmma_f32_16x16x32_bf16(
                  false, a, false, b, (short)0, acc, false, false);
    }

    // ---- epilogue: LDS bounce -> coalesced 16B stores ----
#pragma unroll
    for (int i = 0; i < 8; ++i)
        Ctile[(i + half * 8) * 128 + wave * 16 + ncol] = acc[i];
    __syncthreads();
    {
        int row = threadIdx.x >> 4;          // 0..15
        int seg = (threadIdx.x & 15) * 8;    // 0,8,...,120
        const float4* src = (const float4*)&Ctile[row * 128 + seg];
        float4* dst = (float4*)&Hout[(long long)(rowbase + row) * HCC
                                     + blockIdx.y * 128 + seg];
        dst[0] = src[0];
        dst[1] = src[1];
    }
}

// ---------------- conversion kernels ----------------
__global__ void k_cvt_bf16(const float* __restrict__ src, __bf16* __restrict__ dst,
                           long long n) {
    long long i = (long long)blockIdx.x * blockDim.x + threadIdx.x;
    if (i < n) dst[i] = f2bf(src[i]);
}

// Wt[c*din + k] = bf16(W[k*256 + c])   (transpose + convert, tiny)
__global__ void k_cvt_wt(const float* __restrict__ W, __bf16* __restrict__ Wt, int din) {
    int i = blockIdx.x * blockDim.x + threadIdx.x;
    if (i >= din * HCC) return;
    int k = i / HCC, c = i % HCC;
    Wt[c * din + k] = f2bf(W[i]);
}

// ---------------- small utility kernels ----------------
__global__ void k_fill(float* __restrict__ p, long long n, float v) {
    long long i = (long long)blockIdx.x * blockDim.x + threadIdx.x;
    if (i < n) p[i] = v;
}

__global__ void k_eamean(const float* __restrict__ ea, float* __restrict__ outm) {
    __shared__ float red[256];
    float s = 0.0f;
    for (int i = threadIdx.x; i < EE; i += 256) s += ea[i];
    red[threadIdx.x] = s; __syncthreads();
    for (int off = 128; off > 0; off >>= 1) {
        if (threadIdx.x < off) red[threadIdx.x] += red[threadIdx.x + off];
        __syncthreads();
    }
    if (threadIdx.x == 0) outm[0] = red[0] / (float)EE;
}

// wae[h] = sum_c We[h*C+c]*ae[h*C+c]  (edge-feature attention collapses to ea*wae[h])
__global__ void k_wae(const float* __restrict__ we, const float* __restrict__ ae,
                      float* __restrict__ wae) {
    int h = threadIdx.x;
    if (h < HH) {
        float s = 0.0f;
        for (int c = 0; c < CC; ++c) s += we[h * CC + c] * ae[h * CC + c];
        wae[h] = s;
    }
}

// ss[n,h] = <h[n,h,:], as[h,:]>,  sd likewise
__global__ void k_scores(const float* __restrict__ hlin,
                         const float* __restrict__ as_, const float* __restrict__ ad_,
                         float* __restrict__ ss, float* __restrict__ sd) {
    int i = blockIdx.x * blockDim.x + threadIdx.x;
    if (i >= NN * HH) return;
    int n = i >> 2, h = i & 3;
    const float* row = hlin + (long long)n * HCC + h * CC;
    float s1 = 0.0f, s2 = 0.0f;
    for (int c = 0; c < CC; ++c) {
        float v = row[c];
        s1 += v * as_[h * CC + c];
        s2 += v * ad_[h * CC + c];
    }
    ss[i] = s1; sd[i] = s2;
}

__device__ inline void edge_sd(int e, const int* ei, int& s, int& d) {
    if (e < EE) { s = ei[e]; d = ei[EE + e]; }
    else        { s = d = e - EE; }
}

// alpha = leaky_relu(ss[src]+sd[dst]+ea*wae); running segment max into mx[dst]
__global__ void k_edge_alpha(const int* __restrict__ ei, const float* __restrict__ ea,
                             const float* __restrict__ eam, const float* __restrict__ ss,
                             const float* __restrict__ sd, const float* __restrict__ wae,
                             float* __restrict__ alpha, float* __restrict__ mx) {
    int e = blockIdx.x * blockDim.x + threadIdx.x;
    if (e >= EPQ) return;
    int s, d; edge_sd(e, ei, s, d);
    float eav = (e < EE) ? ea[e] : eam[0];
#pragma unroll
    for (int h = 0; h < HH; ++h) {
        float a = ss[s * HH + h] + sd[d * HH + h] + eav * wae[h];
        a = (a > 0.0f) ? a : 0.2f * a;
        alpha[(long long)e * HH + h] = a;
        atomicMaxF(&mx[d * HH + h], a);
    }
}

// w = exp(alpha - mx[dst]); den[dst] += w  (w overwrites alpha)
__global__ void k_edge_w(const int* __restrict__ ei, float* __restrict__ alpha,
                         const float* __restrict__ mx, float* __restrict__ den) {
    int e = blockIdx.x * blockDim.x + threadIdx.x;
    if (e >= EPQ) return;
    int s, d; edge_sd(e, ei, s, d); (void)s;
#pragma unroll
    for (int h = 0; h < HH; ++h) {
        float w = expf(alpha[(long long)e * HH + h] - mx[d * HH + h]);
        alpha[(long long)e * HH + h] = w;
        atomicAdd(&den[d * HH + h], w);
    }
}

// agg[dst, :] += hlin[src, :] * att[h]   (one block per edge, 256 channels)
__global__ void __launch_bounds__(256)
k_edge_agg(const int* __restrict__ ei, const float* __restrict__ hlin,
           const float* __restrict__ wexp, const float* __restrict__ den,
           float* __restrict__ agg) {
    int e = blockIdx.x;
    int s, d; edge_sd(e, ei, s, d);
    __shared__ float att[HH];
    if (threadIdx.x < HH)
        att[threadIdx.x] = wexp[(long long)e * HH + threadIdx.x] / den[d * HH + threadIdx.x];
    __syncthreads();
    int t = threadIdx.x;
    float v = hlin[(long long)s * HCC + t] * att[t >> 6];
    atomicAdd(&agg[(long long)d * HCC + t], v);
}

// layers 0/1: +bias -> BN(eval) -> ELU, emit bf16 for the next layer's GEMM
__global__ void k_fin01(const float* __restrict__ agg, const float* __restrict__ bias,
                        const float* __restrict__ g, const float* __restrict__ b,
                        const float* __restrict__ m, const float* __restrict__ v,
                        __bf16* __restrict__ out) {
    long long i = (long long)blockIdx.x * blockDim.x + threadIdx.x;
    if (i >= (long long)NN * HCC) return;
    int j = (int)(i & (HCC - 1));
    float x = agg[i] + bias[j];
    float y = (x - m[j]) * rsqrtf(v[j] + 1e-5f) * g[j] + b[j];
    out[i] = f2bf((y > 0.0f) ? y : (expf(y) - 1.0f));
}

// layer 2: head-mean -> +bias -> BN(eval), N x 64 f32 (for pooling)
__global__ void k_fin2(const float* __restrict__ agg, const float* __restrict__ bias,
                       const float* __restrict__ g, const float* __restrict__ b,
                       const float* __restrict__ m, const float* __restrict__ v,
                       float* __restrict__ out) {
    long long i = (long long)blockIdx.x * blockDim.x + threadIdx.x;
    if (i >= (long long)NN * CC) return;
    int n = (int)(i >> 6), c = (int)(i & 63);
    const float* row = agg + (long long)n * HCC + c;
    float x = 0.25f * (row[0] + row[CC] + row[2 * CC] + row[3 * CC]) + bias[c];
    out[i] = (x - m[c]) * rsqrtf(v[c] + 1e-5f) * g[c] + b[c];
}

// global mean/max pooling via atomics
__global__ void k_pool(const float* __restrict__ h, const int* __restrict__ batch,
                       float* __restrict__ gmean, float* __restrict__ gmax,
                       float* __restrict__ gcnt) {
    long long i = (long long)blockIdx.x * blockDim.x + threadIdx.x;
    if (i >= (long long)NN * CC) return;
    int n = (int)(i >> 6), c = (int)(i & 63);
    int gidx = batch[n];
    float v = h[i];
    atomicAdd(&gmean[gidx * CC + c], v);
    atomicMaxF(&gmax[gidx * CC + c], v);
    if (c == 0) atomicAdd(&gcnt[gidx], 1.0f);
}

// per-graph MLP head (tiny): one block per graph
__global__ void __launch_bounds__(128)
k_mlp(const float* __restrict__ gmean, const float* __restrict__ gmax,
      const float* __restrict__ gcnt, const float* __restrict__ emb,
      const int* __restrict__ cids,
      const float* __restrict__ mw1, const float* __restrict__ mb1,
      const float* __restrict__ mw2, const float* __restrict__ mb2,
      const float* __restrict__ mw3, const float* __restrict__ mb3,
      float* __restrict__ out) {
    __shared__ float comb[144];
    __shared__ float z1[128];
    __shared__ float z2[64];
    int gidx = blockIdx.x, t = threadIdx.x;
    float cnt = gcnt[gidx]; if (cnt < 1.0f) cnt = 1.0f;
    if (t < 64) {
        comb[t]      = gmean[gidx * CC + t] / cnt;
        comb[64 + t] = gmax[gidx * CC + t];
    }
    if (t < 16) comb[128 + t] = emb[cids[gidx] * 16 + t];
    __syncthreads();
    {
        float s = mb1[t];
        for (int k = 0; k < 144; ++k) s += comb[k] * mw1[k * 128 + t];
        z1[t] = (s > 0.0f) ? s : 0.0f;
    }
    __syncthreads();
    if (t < 64) {
        float s = mb2[t];
        for (int k = 0; k < 128; ++k) s += z1[k] * mw2[k * 64 + t];
        z2[t] = (s > 0.0f) ? s : 0.0f;
    }
    __syncthreads();
    if (t < 2) {
        float s = mb3[t];
        for (int k = 0; k < 64; ++k) s += z2[k] * mw3[k * 2 + t];
        out[gidx * 2 + t] = s;
    }
}

// =====================================================================
extern "C" void kernel_launch(void* const* d_in, const int* in_sizes, int n_in,
                              void* d_out, int out_size, void* d_ws, size_t ws_size,
                              hipStream_t stream) {
    (void)in_sizes; (void)n_in; (void)out_size; (void)ws_size;
    // ---- inputs (setup_inputs order) ----
    const float* x   = (const float*)d_in[0];
    const float* ea  = (const float*)d_in[1];
    const float* w_[3]  = { (const float*)d_in[2],  (const float*)d_in[7],  (const float*)d_in[12] };
    const float* as_[3] = { (const float*)d_in[3],  (const float*)d_in[8],  (const float*)d_in[13] };
    const float* ad_[3] = { (const float*)d_in[4],  (const float*)d_in[9],  (const float*)d_in[14] };
    const float* we_[3] = { (const float*)d_in[5],  (const float*)d_in[10], (const float*)d_in[15] };
    const float* ae_[3] = { (const float*)d_in[6],  (const float*)d_in[11], (const float*)d_in[16] };
    const float* bias_[3] = { (const float*)d_in[17], (const float*)d_in[18], (const float*)d_in[19] };
    const float* bng[3] = { (const float*)d_in[20], (const float*)d_in[24], (const float*)d_in[28] };
    const float* bnb[3] = { (const float*)d_in[21], (const float*)d_in[25], (const float*)d_in[29] };
    const float* bnm[3] = { (const float*)d_in[22], (const float*)d_in[26], (const float*)d_in[30] };
    const float* bnv[3] = { (const float*)d_in[23], (const float*)d_in[27], (const float*)d_in[31] };
    const float* emb = (const float*)d_in[32];
    const float* mw1 = (const float*)d_in[33]; const float* mb1 = (const float*)d_in[34];
    const float* mw2 = (const float*)d_in[35]; const float* mb2 = (const float*)d_in[36];
    const float* mw3 = (const float*)d_in[37]; const float* mb3 = (const float*)d_in[38];
    const int* ei    = (const int*)d_in[39];
    const int* batch = (const int*)d_in[40];
    const int* cids  = (const int*)d_in[41];
    float* out = (float*)d_out;

    // ---- workspace layout (float-sized slots) ----
    float* w     = (float*)d_ws;
    const long long NHC = (long long)NN * HCC;
    float* hLin  = w;                           // N x 256 f32 (GEMM output)
    float* agg   = hLin + NHC;                  // N x 256 f32 (aggregation)
    float* hA    = agg + NHC;                   // N x 64  f32 (layer-2 output)
    float* ss    = hA + (long long)NN * CC;     // N x H
    float* sd    = ss + (long long)NN * HH;
    float* mx    = sd + (long long)NN * HH;
    float* den   = mx + (long long)NN * HH;
    float* alpha = den + (long long)NN * HH;    // E' x H (reused as exp weights)
    float* eam   = alpha + (long long)EPQ * HH;
    float* waeb  = eam + 1;                     // H
    float* gmean = waeb + HH;                   // G x 64
    float* gmax  = gmean + (long long)GG * CC;  // G x 64
    float* gcnt  = gmax + (long long)GG * CC;   // G
    __bf16* hAb  = (__bf16*)(gcnt + GG);        // N x 256 bf16 (next-layer GEMM input)
    __bf16* xb   = hAb + NHC;                   // N x 32 bf16
    __bf16* Wt   = xb + (long long)NN * INF_;   // 256 x 256 bf16 (transposed weights)

    const float NEGINF = -__builtin_huge_valf();
    const int T = 256;
    const int gridNH  = (NN * HH + T - 1) / T;
    const int gridNHC = (int)((NHC + T - 1) / T);
    const int gridNC  = (NN * CC + T - 1) / T;
    const int gridEP  = (EPQ + T - 1) / T;
    const dim3 gGemm(NN / 16, 2);

    k_eamean<<<1, 256, 0, stream>>>(ea, eam);
    k_cvt_bf16<<<(int)(((long long)NN * INF_ + T - 1) / T), T, 0, stream>>>(
        x, xb, (long long)NN * INF_);

    for (int l = 0; l < 3; ++l) {
        int din = (l == 0) ? INF_ : HCC;
        k_cvt_wt<<<(din * HCC + T - 1) / T, T, 0, stream>>>(w_[l], Wt, din);

        if (l == 0) k_lin_wmma<INF_><<<gGemm, 256, 0, stream>>>(xb,  Wt, hLin);
        else        k_lin_wmma<HCC ><<<gGemm, 256, 0, stream>>>(hAb, Wt, hLin);

        k_scores<<<gridNH, T, 0, stream>>>(hLin, as_[l], ad_[l], ss, sd);
        k_fill<<<gridNH, T, 0, stream>>>(mx, (long long)NN * HH, NEGINF);
        k_fill<<<gridNH, T, 0, stream>>>(den, (long long)NN * HH, 0.0f);
        k_fill<<<gridNHC, T, 0, stream>>>(agg, NHC, 0.0f);
        k_wae<<<1, 32, 0, stream>>>(we_[l], ae_[l], waeb);

        k_edge_alpha<<<gridEP, T, 0, stream>>>(ei, ea, eam, ss, sd, waeb, alpha, mx);
        k_edge_w<<<gridEP, T, 0, stream>>>(ei, alpha, mx, den);
        k_edge_agg<<<EPQ, 256, 0, stream>>>(ei, hLin, alpha, den, agg);

        if (l < 2) k_fin01<<<gridNHC, T, 0, stream>>>(agg, bias_[l], bng[l], bnb[l], bnm[l], bnv[l], hAb);
        else       k_fin2 <<<gridNC,  T, 0, stream>>>(agg, bias_[l], bng[l], bnb[l], bnm[l], bnv[l], hA);
    }

    // ---- pooling + MLP head ----
    k_fill<<<(GG * CC + T - 1) / T, T, 0, stream>>>(gmean, (long long)GG * CC, 0.0f);
    k_fill<<<(GG * CC + T - 1) / T, T, 0, stream>>>(gmax, (long long)GG * CC, NEGINF);
    k_fill<<<1, GG, 0, stream>>>(gcnt, GG, 0.0f);
    k_pool<<<gridNC, T, 0, stream>>>(hA, batch, gmean, gmax, gcnt);
    k_mlp<<<GG, 128, 0, stream>>>(gmean, gmax, gcnt, emb, cids,
                                  mw1, mb1, mw2, mb2, mw3, mb3, out);
}
